// CFPosGenerator_20229295964330
// MI455X (gfx1250) — matrix operand order repaired and checked by reading, not compiled, not run on hardware
//
#include <hip/hip_runtime.h>
#include <hip/hip_bf16.h>

typedef __attribute__((ext_vector_type(16))) __bf16 v16bf;
typedef __attribute__((ext_vector_type(8)))  __bf16 v8bf;
typedef __attribute__((ext_vector_type(4)))  __bf16 v4bf;
typedef __attribute__((ext_vector_type(8)))  float  v8f;

#define RNUM 4
#define MREM 32

// ---------------------------------------------------------------------------
// WMMA fragment helpers (v_wmma_f32_16x16x32_bf16, wave32).
// All fragment gathers are 16B LDS vector loads (ds_load_b128).
__device__ __forceinline__ v16bf cat8(v8bf lo, v8bf hi) {
  return __builtin_shufflevector(lo, hi, 0, 1, 2, 3, 4, 5, 6, 7,
                                 8, 9, 10, 11, 12, 13, 14, 15);
}

// A: 16x32 bf16, row-major LDS, stride lda (multiple of 8).
// lanes 0-15: M=lane,   elems = K k0+0..7  | k0+16..23
// lanes16-31: M=lane-16,elems = K k0+8..15 | k0+24..31
__device__ __forceinline__ v16bf load_a_frag(const __bf16* A, int lda, int lane, int k0) {
  int m   = lane & 15;
  int klo = k0 + ((lane & 16) ? 8  : 0);
  int khi = k0 + ((lane & 16) ? 24 : 16);
  const __bf16* row = A + m * lda;
  v8bf lo = *(const v8bf*)(row + klo);
  v8bf hi = *(const v8bf*)(row + khi);
  return cat8(lo, hi);
}

// B: 32x16 bf16 (KxN) taken from an N-major (transposed) weight tile
// Wt[n*ldk + k]; lanes 0-15: N=lane, K=k0+0..7 | k0+16..23; lanes 16-31: N=lane-16, K=k0+8..15 | k0+24..31.
__device__ __forceinline__ v16bf load_b_fragT(const __bf16* Wt, int ldk, int lane,
                                              int k0, int n0) {
  int n  = n0 + (lane & 15);
  int kb = k0 + ((lane & 16) ? 8 : 0);
  const __bf16* col = Wt + n * ldk + kb;
  v8bf lo = *(const v8bf*)(col);
  v8bf hi = *(const v8bf*)(col + 16);
  return cat8(lo, hi);
}

// C/D 16x16 f32: VGPR r -> M = r + ((lane<16)?0:8), N = lane&15
__device__ __forceinline__ void store_cd(float* Dst, int ldd, int lane, int n0, v8f c) {
  int n  = n0 + (lane & 15);
  int mb = (lane & 16) ? 8 : 0;
#pragma unroll
  for (int r = 0; r < 8; ++r) Dst[(mb + r) * ldd + n] = c[r];
}

__device__ __forceinline__ v8f wmma_bf16(v16bf a, v16bf b, v8f c) {
  return __builtin_amdgcn_wmma_f32_16x16x32_bf16(false, a, false, b, (short)0, c,
                                                 false, false);
}

__device__ __forceinline__ v4bf to_bf4(float4 v) {
  v4bf r;
  r[0] = (__bf16)v.x; r[1] = (__bf16)v.y; r[2] = (__bf16)v.z; r[3] = (__bf16)v.w;
  return r;
}

// ---------------------------------------------------------------------------
// Kernel 0: weights -> TRANSPOSED bf16 (N-major) in ws; seed cf_kg1/cf_kg2.
__global__ void prep_kernel(const float* uiW, const float* e1W, const float* e2W,
                            const int* kg, int NK,
                            __bf16* uiWt, __bf16* e1Wt, __bf16* e2Wt,
                            int* cf1, int* cf2) {
  int i = blockIdx.x * blockDim.x + threadIdx.x;
  int stride = gridDim.x * blockDim.x;
  for (int t = i; t < 64 * 128; t += stride) {       // uiWt[n*128+k] = uiW[k*64+n]
    int n = t >> 7, k = t & 127;
    uiWt[t] = (__bf16)uiW[k * 64 + n];
  }
  for (int t = i; t < 64 * 64; t += stride) {        // e*Wt[n*64+k] = e*W[k*64+n]
    int n = t >> 6, k = t & 63;
    e1Wt[t] = (__bf16)e1W[k * 64 + n];
    e2Wt[t] = (__bf16)e2W[k * 64 + n];
  }
  for (int t = i; t < NK; t += stride) {
    int v = kg[t];
    cf1[t] = v;
    cf2[t] = v;
  }
}

// ---------------------------------------------------------------------------
// Kernel 1: ui_e = l2norm(concat(users_e, item_e) @ ui_W + ui_b)
// 128 threads = 4 waves; each wave one 16x64 tile = 16 WMMAs (4 N x 4 K).
__global__ void __launch_bounds__(128) ui_kernel(
    const int* users, const float* user_emb, const float* item_emb,
    const __bf16* uiWt, const float* ui_b,
    float* ui_e, float* users_e, int B) {
  __shared__ alignas(16) __bf16 Wl[64 * 128];     // transposed [n][k], ldk=128
  __shared__ alignas(16) __bf16 Al[4][16 * 128];
  __shared__ alignas(16) float  Dl[4][16 * 64];
  __shared__ float bias[64];
  __shared__ float scale[4][16];

  int tid = threadIdx.x, wave = tid >> 5, lane = tid & 31;
  int bm0 = blockIdx.x * 64 + wave * 16;

  {  // 16KB weight tile, 16B-wide copies
    const uint4* src = (const uint4*)uiWt;
    uint4* dst = (uint4*)Wl;
    for (int t = tid; t < (64 * 128 * 2) / 16; t += 128) dst[t] = src[t];
  }
  if (tid < 64) bias[tid] = ui_b[tid];

  // stage A: 16 rows x 128 cols, float4 gathers -> v4bf LDS stores
  for (int t = lane; t < 16 * 32; t += 32) {
    int m = t >> 5, c = t & 31, k = c * 4;
    long b = bm0 + m;
    float4 v;
    if (k < 64) {
      v = ((const float4*)(user_emb + (long)users[b] * 64))[c];
      ((float4*)(users_e + b * 64))[c] = v;   // cache users_e f32 for step 2
    } else {
      v = ((const float4*)(item_emb + b * 64))[c - 16];
    }
    *(v4bf*)&Al[wave][m * 128 + k] = to_bf4(v);
  }
  __syncthreads();

#pragma unroll
  for (int nt = 0; nt < 4; ++nt) {
    v8f acc = {};
#pragma unroll
    for (int kt = 0; kt < 4; ++kt) {
      v16bf a  = load_a_frag(&Al[wave][0], 128, lane, kt * 32);
      v16bf bb = load_b_fragT(Wl, 128, lane, kt * 32, nt * 16);
      acc = wmma_bf16(a, bb, acc);
    }
    store_cd(&Dl[wave][0], 64, lane, nt * 16, acc);
  }
  __syncthreads();

  if (lane < 16) {  // bias + row sum-of-squares, float4-wide
    float4* row = (float4*)&Dl[wave][lane * 64];
    const float4* b4 = (const float4*)bias;
    float ss = 0.f;
#pragma unroll
    for (int i = 0; i < 16; ++i) {
      float4 v = row[i], bb = b4[i];
      v.x += bb.x; v.y += bb.y; v.z += bb.z; v.w += bb.w;
      ss += v.x * v.x + v.y * v.y + v.z * v.z + v.w * v.w;
      row[i] = v;
    }
    scale[wave][lane] = 1.0f / fmaxf(sqrtf(ss), 1e-12f);
  }
  __syncthreads();
  for (int t = lane; t < 16 * 16; t += 32) {
    int m = t >> 4, c = t & 15;
    float s = scale[wave][m];
    float4 v = ((float4*)&Dl[wave][m * 64])[c];
    v.x *= s; v.y *= s; v.z *= s; v.w *= s;
    ((float4*)(ui_e + (long)(bm0 + m) * 64))[c] = v;
  }
}

// ---------------------------------------------------------------------------
// Kernel 2: per item b: ne = l2norm(E[nb] @ e1_W + b); score vs ui_e;
// softmax + top-4; emit idx/rel/replaced_e/ap1; zero cf_kg1 slots.
__global__ void __launch_bounds__(128) step1_kernel(
    const int* items, const int* kg, const int* nrel,
    const float* ent_emb, const __bf16* e1Wt, const float* e1_b,
    const float* ui_e,
    int* cf1, float* ap1_out,
    int* s1_idx_ws, int* sel_rel_ws, float* repl_e_ws) {
  __shared__ alignas(16) __bf16 Wl[64 * 64];      // transposed [n][k], ldk=64
  __shared__ alignas(16) __bf16 Al[4][16 * 64];
  __shared__ alignas(16) float  NE[64 * 64];
  __shared__ int    nb[64];
  __shared__ float  uie[64];
  __shared__ float  bias[64];
  __shared__ float  sc[64];
  __shared__ int    top_idx[RNUM];

  int tid = threadIdx.x, wave = tid >> 5, lane = tid & 31;
  int b = blockIdx.x;
  int item = items[b];

  {
    const uint4* src = (const uint4*)e1Wt;
    uint4* dst = (uint4*)Wl;
    for (int t = tid; t < (64 * 64 * 2) / 16; t += 128) dst[t] = src[t];
  }
  if (tid < 64) {
    nb[tid]   = kg[(long)item * 64 + tid];
    uie[tid]  = ui_e[(long)b * 64 + tid];
    bias[tid] = e1_b[tid];
  }
  __syncthreads();

  int m0 = wave * 16;
  for (int t = lane; t < 16 * 16; t += 32) {      // stage 16 rows x 64 cols
    int m = t >> 4, c = t & 15;
    float4 v = ((const float4*)(ent_emb + (long)nb[m0 + m] * 64))[c];
    *(v4bf*)&Al[wave][m * 64 + c * 4] = to_bf4(v);
  }
  __syncthreads();

#pragma unroll
  for (int nt = 0; nt < 4; ++nt) {
    v8f acc = {};
#pragma unroll
    for (int kt = 0; kt < 2; ++kt) {
      v16bf a  = load_a_frag(&Al[wave][0], 64, lane, kt * 32);
      v16bf bb = load_b_fragT(Wl, 64, lane, kt * 32, nt * 16);
      acc = wmma_bf16(a, bb, acc);
    }
    store_cd(&NE[m0 * 64], 64, lane, nt * 16, acc);
  }
  __syncthreads();

  if (tid < 64) {  // bias + l2norm + dot(ui_e), float4-wide
    float4* row = (float4*)&NE[tid * 64];
    const float4* b4 = (const float4*)bias;
    const float4* u4 = (const float4*)uie;
    float ss = 0.f;
#pragma unroll
    for (int i = 0; i < 16; ++i) {
      float4 v = row[i], bb = b4[i];
      v.x += bb.x; v.y += bb.y; v.z += bb.z; v.w += bb.w;
      ss += v.x * v.x + v.y * v.y + v.z * v.z + v.w * v.w;
      row[i] = v;
    }
    float s = 1.f / fmaxf(sqrtf(ss), 1e-12f);
    float dot = 0.f;
#pragma unroll
    for (int i = 0; i < 16; ++i) {
      float4 v = row[i], u = u4[i];
      v.x *= s; v.y *= s; v.z *= s; v.w *= s;
      row[i] = v;
      dot += v.x * u.x + v.y * u.y + v.z * u.z + v.w * u.w;
    }
    sc[tid] = dot;
  }
  __syncthreads();

  if (tid == 0) {  // softmax over 64 + top-4 (tiny, serial)
    float mx = sc[0];
    for (int i = 1; i < 64; ++i) mx = fmaxf(mx, sc[i]);
    float sum = 0.f;
    for (int i = 0; i < 64; ++i) sum += __expf(sc[i] - mx);
    unsigned long long taken = 0ull;
    float psum = 0.f;
    for (int r = 0; r < RNUM; ++r) {
      int best = 0; float bv = -3.0e38f;
      for (int i = 0; i < 64; ++i) {
        if ((taken >> i) & 1ull) continue;
        if (sc[i] > bv) { bv = sc[i]; best = i; }
      }
      taken |= 1ull << best;
      top_idx[r] = best;
      psum += __expf(sc[best] - mx) / sum;
    }
    ap1_out[b] = __logf(psum * (1.0f / RNUM));
  }
  __syncthreads();

  if (tid < RNUM) {
    int idx = top_idx[tid];
    s1_idx_ws[b * RNUM + tid]  = idx;
    sel_rel_ws[b * RNUM + tid] = nrel[(long)item * 64 + idx];
    cf1[(long)item * 64 + idx] = 0;
  }
  for (int t = tid; t < RNUM * 16; t += 128) {     // save normalized replaced rows
    int r = t >> 4, c = t & 15;
    float4 v = ((const float4*)&NE[top_idx[r] * 64])[c];
    ((float4*)(repl_e_ws + ((long)b * RNUM + r) * 64))[c] = v;
  }
}

// ---------------------------------------------------------------------------
// Kernel 3: per (b,r): score 64 candidates vs users_e, keep top-32,
// CE = l2norm(E[sel] @ e2_W + b), s2 = CE @ (ui_e*replaced_e), sample.
__global__ void __launch_bounds__(64) step2_kernel(
    const int* items, const int* all_cands, const float* ent_emb,
    const __bf16* e2Wt, const float* e2_b,
    const float* ui_e, const float* users_e, const float* repl_e,
    const int* s1_idx_ws, const int* sel_rel_ws,
    int* cf2, float* p2_ws) {
  __shared__ alignas(16) __bf16 Wl[64 * 64];      // transposed [n][k], ldk=64
  __shared__ alignas(16) __bf16 Al[32 * 64];
  __shared__ alignas(16) float  CE[32 * 64];
  __shared__ int    cand[64];
  __shared__ float  sc[64];
  __shared__ int    sel[MREM];
  __shared__ alignas(16) float q[64];
  __shared__ alignas(16) float bias[64];
  __shared__ float  s2[MREM];

  int tid = threadIdx.x, wave = tid >> 5, lane = tid & 31;
  long br = blockIdx.x;
  int b = (int)(br >> 2);

  {
    const uint4* src = (const uint4*)e2Wt;
    uint4* dst = (uint4*)Wl;
    for (int t = tid; t < (64 * 64 * 2) / 16; t += 64) dst[t] = src[t];
  }
  int rel = sel_rel_ws[br];
  cand[tid] = all_cands[(long)rel * 64 + tid];
  q[tid]    = ui_e[(long)b * 64 + tid] * repl_e[br * 64 + tid];
  bias[tid] = e2_b[tid];
  __syncthreads();

  {  // candidate scoring: thread c dots users_e[b] with entity[cand[c]]
    const float4* u4 = (const float4*)(users_e + (long)b * 64);
    const float4* e4 = (const float4*)(ent_emb + (long)cand[tid] * 64);
    float dot = 0.f;
#pragma unroll
    for (int i = 0; i < 16; ++i) {
      float4 u = u4[i], e = e4[i];
      dot += u.x * e.x + u.y * e.y + u.z * e.z + u.w * e.w;
    }
    sc[tid] = dot;
  }
  __syncthreads();
  {  // rank-compact top-32 (descending, index tiebreak == top_k order)
    float v = sc[tid];
    int rank = 0;
    for (int j = 0; j < 64; ++j)
      rank += (sc[j] > v) || (sc[j] == v && j < tid);
    if (rank < MREM) sel[rank] = cand[tid];
  }
  __syncthreads();

  for (int t = tid; t < MREM * 16; t += 64) {     // stage 32 rows x 64 cols
    int m = t >> 4, c = t & 15;
    float4 v = ((const float4*)(ent_emb + (long)sel[m] * 64))[c];
    *(v4bf*)&Al[m * 64 + c * 4] = to_bf4(v);
  }
  __syncthreads();

  int m0 = wave * 16;
#pragma unroll
  for (int nt = 0; nt < 4; ++nt) {
    v8f acc = {};
#pragma unroll
    for (int kt = 0; kt < 2; ++kt) {
      v16bf a  = load_a_frag(&Al[m0 * 64], 64, lane, kt * 32);
      v16bf bb = load_b_fragT(Wl, 64, lane, kt * 32, nt * 16);
      acc = wmma_bf16(a, bb, acc);
    }
    store_cd(&CE[m0 * 64], 64, lane, nt * 16, acc);
  }
  __syncthreads();

  if (tid < MREM) {  // bias + l2norm + dot(q), float4-wide
    float4* row = (float4*)&CE[tid * 64];
    const float4* b4 = (const float4*)bias;
    const float4* q4 = (const float4*)q;
    float ss = 0.f;
#pragma unroll
    for (int i = 0; i < 16; ++i) {
      float4 v = row[i], bb = b4[i];
      v.x += bb.x; v.y += bb.y; v.z += bb.z; v.w += bb.w;
      ss += v.x * v.x + v.y * v.y + v.z * v.z + v.w * v.w;
      row[i] = v;
    }
    float s = 1.f / fmaxf(sqrtf(ss), 1e-12f);
    float dot = 0.f;
#pragma unroll
    for (int i = 0; i < 16; ++i) {
      float4 v = row[i], qq = q4[i];
      dot += (v.x * qq.x + v.y * qq.y + v.z * qq.z + v.w * qq.w);
    }
    s2[tid] = dot * s;
  }
  __syncthreads();

  if (tid == 0) {  // log-softmax + categorical sample (hash RNG, deterministic)
    float mx = s2[0];
    for (int i = 1; i < MREM; ++i) mx = fmaxf(mx, s2[i]);
    float sum = 0.f;
    for (int i = 0; i < MREM; ++i) sum += __expf(s2[i] - mx);
    unsigned long long x = (unsigned long long)br * 0x9E3779B97F4A7C15ull + 42ull;
    x ^= x >> 33; x *= 0xFF51AFD7ED558CCDull;
    x ^= x >> 33; x *= 0xC4CEB9FE1A85EC53ull; x ^= x >> 33;
    float u = (float)(x >> 40) * (1.0f / 16777216.0f);
    float cdf = 0.f; int pick = MREM - 1;
    for (int i = 0; i < MREM; ++i) {
      float p = __expf(s2[i] - mx) / sum;
      cdf += p;
      if (u < cdf) { pick = i; break; }
    }
    p2_ws[br] = __expf(s2[pick] - mx) / sum;
    cf2[(long)items[b] * 64 + s1_idx_ws[br]] = sel[pick];
  }
}

// ---------------------------------------------------------------------------
// Kernel 4: ap2[b] = ap1[b] + log(mean_r step2_prob)
__global__ void ap2_kernel(const float* ap1, const float* p2, float* ap2, int B) {
  int b = blockIdx.x * blockDim.x + threadIdx.x;
  if (b < B) {
    float m = 0.25f * (p2[b * 4] + p2[b * 4 + 1] + p2[b * 4 + 2] + p2[b * 4 + 3]);
    ap2[b] = ap1[b] + __logf(m);
  }
}

// ---------------------------------------------------------------------------
extern "C" void kernel_launch(void* const* d_in, const int* in_sizes, int n_in,
                              void* d_out, int out_size, void* d_ws, size_t ws_size,
                              hipStream_t stream) {
  const int*   users     = (const int*)  d_in[0];
  const int*   items     = (const int*)  d_in[1];
  const int*   kg        = (const int*)  d_in[2];
  const int*   cands     = (const int*)  d_in[3];
  const int*   nrel      = (const int*)  d_in[4];
  const float* user_emb  = (const float*)d_in[5];
  const float* ent_emb   = (const float*)d_in[6];
  const float* item_emb  = (const float*)d_in[7];
  const float* uiW       = (const float*)d_in[8];
  const float* ui_b      = (const float*)d_in[9];
  const float* e1W       = (const float*)d_in[10];
  const float* e1_b      = (const float*)d_in[11];
  const float* e2W       = (const float*)d_in[12];
  const float* e2_b      = (const float*)d_in[13];

  const int B  = in_sizes[0];       // 4096
  const int NK = in_sizes[2];       // n_items * 64

  // Output layout: cf_kg1[NK] (int bits), ap1[B], cf_kg2[NK] (int bits), ap2[B]
  float* out  = (float*)d_out;
  int*   cf1  = (int*)out;
  float* ap1  = out + NK;
  int*   cf2  = (int*)(out + NK + B);
  float* ap2  = out + 2 * (long)NK + B;

  // Workspace layout
  char* w = (char*)d_ws;
  __bf16* uiWt   = (__bf16*)w; w += 64 * 128 * sizeof(__bf16);
  __bf16* e1Wt   = (__bf16*)w; w += 64 * 64 * sizeof(__bf16);
  __bf16* e2Wt   = (__bf16*)w; w += 64 * 64 * sizeof(__bf16);
  float* ui_e    = (float*)w;  w += (long)B * 64 * sizeof(float);
  float* users_e = (float*)w;  w += (long)B * 64 * sizeof(float);
  int*   s1_idx  = (int*)w;    w += (long)B * RNUM * sizeof(int);
  int*   sel_rel = (int*)w;    w += (long)B * RNUM * sizeof(int);
  float* repl_e  = (float*)w;  w += (long)B * RNUM * 64 * sizeof(float);
  float* p2      = (float*)w;  w += (long)B * RNUM * sizeof(float);
  (void)ws_size; (void)n_in; (void)out_size;

  prep_kernel<<<1024, 256, 0, stream>>>(uiW, e1W, e2W, kg, NK,
                                        uiWt, e1Wt, e2Wt, cf1, cf2);

  ui_kernel<<<B / 64, 128, 0, stream>>>(users, user_emb, item_emb, uiWt, ui_b,
                                        ui_e, users_e, B);

  step1_kernel<<<B, 128, 0, stream>>>(items, kg, nrel, ent_emb, e1Wt, e1_b,
                                      ui_e, cf1, ap1, s1_idx, sel_rel, repl_e);

  step2_kernel<<<B * RNUM, 64, 0, stream>>>(items, cands, ent_emb, e2Wt, e2_b,
                                            ui_e, users_e, repl_e, s1_idx, sel_rel,
                                            cf2, p2);

  ap2_kernel<<<(B + 255) / 256, 256, 0, stream>>>(ap1, p2, ap2, B);
}
